// SPTransformer_80668075753646
// MI455X (gfx1250) — matrix operand order repaired and checked by reading, not compiled, not run on hardware
//
#include <hip/hip_runtime.h>
#include <hip/hip_bf16.h>
#include <math.h>

#define Nn   50000
#define Ee   640000
#define Dd   128
#define Hh   8
#define DHd  16
#define FFf  256
#define Cc   5
#define Ll   6
#define EPSf 1e-5f
#define KMAX 256

typedef __attribute__((ext_vector_type(16))) _Float16 v16h;
typedef __attribute__((ext_vector_type(8)))  float    v8f;
typedef __attribute__((ext_vector_type(4)))  int      v4i;

#if defined(__has_builtin)
#  if __has_builtin(__builtin_amdgcn_global_load_async_to_lds_b128) && \
      __has_builtin(__builtin_amdgcn_s_wait_asynccnt)
#    define HAS_ASYNC_LDS 1
#  else
#    define HAS_ASYNC_LDS 0
#  endif
#else
#  define HAS_ASYNC_LDS 0
#endif

__device__ __forceinline__ void atomAddF(float* p, float v) {
    unsafeAtomicAdd(p, v);   // native global_atomic_add_f32 on gfx1250
}

__device__ __forceinline__ void atomMaxF(float* addr, float val) {
    // monotonic int-punning float max; buffer initialized to -inf
    if (val >= 0.0f) atomicMax((int*)addr, __float_as_int(val));
    else             atomicMin((unsigned int*)addr, (unsigned int)__float_as_int(val));
}

// ---------------------------------------------------------------------------
// WMMA GEMM: C[M,Nc] = A[M,K] @ B[K,Nc] (+bias, +residual, relu per flags)
// block = 256 thr = 8 waves; block tile 256(M) x 16(N); wave tile 32x16
// (2 x v_wmma_f32_16x16x32_f16 per K-step sharing one B fragment).
// Full B panel (K x 16) staged in LDS ONCE (async-DMA path when available),
// so the K loop is barrier-free.
// flags: 1 = add bias[Nc], 2 = relu, 4 = add residual[M,Nc]
// ---------------------------------------------------------------------------
__global__ __launch_bounds__(256)
void k_gemm_wmma(const float* __restrict__ A, const float* __restrict__ B,
                 const float* __restrict__ bias, const float* __restrict__ resid,
                 float* __restrict__ Cout, int M, int K, int Nc, int flags) {
    __shared__ __attribute__((aligned(64))) _Float16 Bs[16][KMAX];   // [col][k] f16
#if HAS_ASYNC_LDS
    __shared__ __attribute__((aligned(64))) float    BsF[KMAX][16];  // [k][col] f32
#endif

    const int tid     = threadIdx.x;
    const int lane    = tid & 31;
    const int wave    = tid >> 5;
    const int n0      = blockIdx.x * 16;
    const int m0w     = blockIdx.y * 256 + wave * 32;   // wave owns rows m0w..m0w+31
    const int halfsel = lane >> 4;       // 0: lanes 0-15, 1: lanes 16-31
    const int l15     = lane & 15;
    const int kbaseA  = halfsel * 16;    // A frag: K {0..15} / {16..31}
    const int kbaseB  = halfsel * 16;    // B frag: K {0..15} / {16..31}
    const int arow0   = min(m0w + l15,      M - 1);
    const int arow1   = min(m0w + 16 + l15, M - 1);

    // ---- stage full B panel (K x 16) into LDS as f16, once ----
#if HAS_ASYNC_LDS
    // async-DMA the f32 panel into LDS (16B chunks, per-lane addressed),
    // then convert LDS f32 -> LDS f16.
    for (int c = tid; c < K * 4; c += 256) {
        int kk = c >> 2, cg = (c & 3) * 4;
        __builtin_amdgcn_global_load_async_to_lds_b128(
            (__attribute__((address_space(1))) v4i*)(B + (size_t)kk * Nc + n0 + cg),
            (__attribute__((address_space(3))) v4i*)&BsF[kk][cg], 0, 0);
    }
    __builtin_amdgcn_s_wait_asynccnt(0);
    __syncthreads();
    for (int e = tid; e < K * 16; e += 256) {
        int col = e & 15, kk = e >> 4;
        Bs[col][kk] = (_Float16)BsF[kk][col];
    }
#else
    for (int e = tid; e < K * 16; e += 256) {
        int col = e & 15, kk = e >> 4;
        Bs[col][kk] = (_Float16)B[(size_t)kk * Nc + (n0 + col)];
    }
#endif
    __syncthreads();

    v8f acc0 = {}, acc1 = {};

    for (int k0 = 0; k0 < K; k0 += 32) {
        // ---- A fragments: one contiguous 16-float run per lane per row ----
        const float* arp0 = A + (size_t)arow0 * K + k0 + kbaseA;
        const float* arp1 = A + (size_t)arow1 * K + k0 + kbaseA;
        __builtin_prefetch(arp0 + 32, 0, 0);   // next K-slice -> global_prefetch
        float4 a0 = *(const float4*)(arp0 + 0);
        float4 a1 = *(const float4*)(arp0 + 4);
        float4 a2 = *(const float4*)(arp0 + 8);
        float4 a3 = *(const float4*)(arp0 + 12);
        float4 c0 = *(const float4*)(arp1 + 0);
        float4 c1 = *(const float4*)(arp1 + 4);
        float4 c2 = *(const float4*)(arp1 + 8);
        float4 c3 = *(const float4*)(arp1 + 12);
        v16h af0, af1;
        af0[0]  = (_Float16)a0.x; af0[1]  = (_Float16)a0.y;
        af0[2]  = (_Float16)a0.z; af0[3]  = (_Float16)a0.w;
        af0[4]  = (_Float16)a1.x; af0[5]  = (_Float16)a1.y;
        af0[6]  = (_Float16)a1.z; af0[7]  = (_Float16)a1.w;
        af0[8]  = (_Float16)a2.x; af0[9]  = (_Float16)a2.y;
        af0[10] = (_Float16)a2.z; af0[11] = (_Float16)a2.w;
        af0[12] = (_Float16)a3.x; af0[13] = (_Float16)a3.y;
        af0[14] = (_Float16)a3.z; af0[15] = (_Float16)a3.w;
        af1[0]  = (_Float16)c0.x; af1[1]  = (_Float16)c0.y;
        af1[2]  = (_Float16)c0.z; af1[3]  = (_Float16)c0.w;
        af1[4]  = (_Float16)c1.x; af1[5]  = (_Float16)c1.y;
        af1[6]  = (_Float16)c1.z; af1[7]  = (_Float16)c1.w;
        af1[8]  = (_Float16)c2.x; af1[9]  = (_Float16)c2.y;
        af1[10] = (_Float16)c2.z; af1[11] = (_Float16)c2.w;
        af1[12] = (_Float16)c3.x; af1[13] = (_Float16)c3.y;
        af1[14] = (_Float16)c3.z; af1[15] = (_Float16)c3.w;

        // ---- B fragment: 16 contiguous halves from LDS, shared by both wmmas ----
        v16h bf = *(const v16h*)&Bs[l15][k0 + kbaseB];

        acc0 = __builtin_amdgcn_wmma_f32_16x16x32_f16(
                   false, af0, false, bf, (short)0, acc0, false, false);
        acc1 = __builtin_amdgcn_wmma_f32_16x16x32_f16(
                   false, af1, false, bf, (short)0, acc1, false, false);
    }

    // ---- epilogue: C layout — VGPR r: lanes0-15 M=r, lanes16-31 M=r+8 ----
    const int col = n0 + l15;
    const float bv = (flags & 1) ? bias[col] : 0.0f;

    auto emit = [&](int row, float v) {
        v += bv;
        if (flags & 4) v += resid[(size_t)row * Nc + col];
        if (flags & 2) v = fmaxf(v, 0.0f);
        Cout[(size_t)row * Nc + col] = v;
    };

    if (m0w + 32 <= M) {          // fast path: whole wave tile in bounds
#pragma unroll
        for (int r = 0; r < 8; ++r) {
            int row = m0w + r + halfsel * 8;
            emit(row,      acc0[r]);
            emit(row + 16, acc1[r]);
        }
    } else {                      // boundary block only
#pragma unroll
        for (int r = 0; r < 8; ++r) {
            int row = m0w + r + halfsel * 8;
            if (row < M)      emit(row,      acc0[r]);
            if (row + 16 < M) emit(row + 16, acc1[r]);
        }
    }
}

// ---------------------------------------------------------------------------
// misc elementwise / attention kernels
// ---------------------------------------------------------------------------
__global__ void k_fill(float* __restrict__ p, float val, int n) {
    int i = blockIdx.x * blockDim.x + threadIdx.x;
    if (i < n) p[i] = val;
}

// per-(edge,head) attention score: s = <q[dst,h], k[src,h]> / sqrt(16)
__global__ void k_scores(const float* __restrict__ q, const float* __restrict__ kk,
                         const int* __restrict__ src, const int* __restrict__ dst,
                         float* __restrict__ sco) {
    int idx = blockIdx.x * blockDim.x + threadIdx.x;
    if (idx >= Ee * Hh) return;
    int e = idx >> 3, h = idx & 7;
    const float4* qp = (const float4*)(q  + (size_t)dst[e] * Dd + h * DHd);
    const float4* kp = (const float4*)(kk + (size_t)src[e] * Dd + h * DHd);
    float dot = 0.0f;
#pragma unroll
    for (int j = 0; j < 4; ++j) {
        float4 a = qp[j], b = kp[j];
        dot += a.x * b.x + a.y * b.y + a.z * b.z + a.w * b.w;
    }
    sco[idx] = dot * 0.25f;   // 1/sqrt(16)
}

__global__ void k_segmax(const float* __restrict__ sco, const int* __restrict__ dst,
                         float* __restrict__ smax) {
    int idx = blockIdx.x * blockDim.x + threadIdx.x;
    if (idx >= Ee * Hh) return;
    int e = idx >> 3, h = idx & 7;
    atomMaxF(&smax[(size_t)dst[e] * Hh + h], sco[idx]);
}

// guard isolated nodes (-inf -> 0) and zero the denominator
__global__ void k_fixmax_zeroden(float* __restrict__ smax, float* __restrict__ den) {
    int i = blockIdx.x * blockDim.x + threadIdx.x;
    if (i >= Nn * Hh) return;
    float m = smax[i];
    if (!(m >= -3.0e38f)) m = 0.0f;
    smax[i] = m;
    den[i]  = 0.0f;
}

__global__ void k_expsum(const float* __restrict__ sco, const int* __restrict__ dst,
                         const float* __restrict__ smax,
                         float* __restrict__ ee, float* __restrict__ den) {
    int idx = blockIdx.x * blockDim.x + threadIdx.x;
    if (idx >= Ee * Hh) return;
    int e = idx >> 3, h = idx & 7;
    int d = dst[e];
    float ev = __expf(sco[idx] - smax[(size_t)d * Hh + h]);
    ee[idx] = ev;
    atomAddF(&den[(size_t)d * Hh + h], ev);
}

// per-(edge,dim) aggregation: att[dst,d] += alpha * v[src,d]
__global__ void k_aggregate(const float* __restrict__ ee, const float* __restrict__ den,
                            const float* __restrict__ vv,
                            const int* __restrict__ src, const int* __restrict__ dst,
                            float* __restrict__ att) {
    int idx = blockIdx.x * blockDim.x + threadIdx.x;
    if (idx >= Ee * Dd) return;
    int e = idx >> 7, d = idx & 127;
    int h = d >> 4;
    int sn = src[e], dn = dst[e];
    float dnm   = den[(size_t)dn * Hh + h];
    float alpha = ee[(size_t)e * Hh + h] / (dnm > 0.0f ? dnm : 1.0f);
    atomAddF(&att[(size_t)dn * Dd + d], alpha * vv[(size_t)sn * Dd + d]);
}

// BatchNorm stage 1: column partial sums (block-local regs -> global atomics)
__global__ __launch_bounds__(128)
void k_bn_stats(const float* __restrict__ x, float* __restrict__ csum,
                float* __restrict__ csq) {
    int col = threadIdx.x;   // 0..127
    float s = 0.0f, q = 0.0f;
    for (int r = blockIdx.x; r < Nn; r += gridDim.x) {
        float v = x[(size_t)r * Dd + col];
        s += v; q += v * v;
    }
    atomAddF(&csum[col], s);
    atomAddF(&csq[col],  q);
}

// BatchNorm stage 2: normalize + affine
__global__ void k_bn_apply(const float* __restrict__ x, const float* __restrict__ csum,
                           const float* __restrict__ csq, const float* __restrict__ g,
                           const float* __restrict__ b, float* __restrict__ out) {
    int i = blockIdx.x * blockDim.x + threadIdx.x;
    if (i >= Nn * Dd) return;
    int col = i & 127;
    const float invN = 1.0f / (float)Nn;
    float mean = csum[col] * invN;
    float var  = csq[col] * invN - mean * mean;
    out[i] = (x[i] - mean) * rsqrtf(var + EPSf) * g[col] + b[col];
}

// final head: [N,128] @ [128,5] + bias
__global__ void k_head(const float* __restrict__ x, const float* __restrict__ Wp,
                       const float* __restrict__ bp, float* __restrict__ out) {
    int n = blockIdx.x * blockDim.x + threadIdx.x;
    if (n >= Nn) return;
    float acc[Cc];
#pragma unroll
    for (int c = 0; c < Cc; ++c) acc[c] = bp[c];
    const float* xr = x + (size_t)n * Dd;
    for (int d = 0; d < Dd; ++d) {
        float xv = xr[d];
#pragma unroll
        for (int c = 0; c < Cc; ++c) acc[c] += xv * Wp[d * Cc + c];
    }
#pragma unroll
    for (int c = 0; c < Cc; ++c) out[(size_t)n * Cc + c] = acc[c];
}

// ---------------------------------------------------------------------------
// host orchestration
// ---------------------------------------------------------------------------
static inline void launch_gemm(const float* A, const float* B, const float* bias,
                               const float* resid, float* Cout,
                               int M, int K, int Nc, int flags, hipStream_t s) {
    dim3 grid(Nc / 16, (M + 255) / 256);
    hipLaunchKernelGGL(k_gemm_wmma, grid, dim3(256), 0, s,
                       A, B, bias, resid, Cout, M, K, Nc, flags);
}

extern "C" void kernel_launch(void* const* d_in, const int* in_sizes, int n_in,
                              void* d_out, int out_size, void* d_ws, size_t ws_size,
                              hipStream_t stream) {
    (void)in_sizes; (void)n_in; (void)out_size; (void)ws_size;

    const float* x_in = (const float*)d_in[0];
    const int*   ei   = (const int*)d_in[1];
    const int*   src  = ei;
    const int*   dst  = ei + Ee;
    const float* Wq   = (const float*)d_in[2];
    const float* Wk   = (const float*)d_in[3];
    const float* Wv   = (const float*)d_in[4];
    const float* Wo   = (const float*)d_in[5];
    const float* g1   = (const float*)d_in[6];
    const float* bn1  = (const float*)d_in[7];
    const float* W1   = (const float*)d_in[8];
    const float* bf1  = (const float*)d_in[9];
    const float* W2   = (const float*)d_in[10];
    const float* bf2  = (const float*)d_in[11];
    const float* g2   = (const float*)d_in[12];
    const float* bn2  = (const float*)d_in[13];
    const float* Wp   = (const float*)d_in[14];
    const float* bp   = (const float*)d_in[15];
    float*       out  = (float*)d_out;

    // carve workspace (all chunk sizes are multiples of 16 floats -> 64B aligned)
    float* w = (float*)d_ws;
    size_t off = 0;
    float* xb   = w + off; off += (size_t)Nn * Dd;
    float* qb   = w + off; off += (size_t)Nn * Dd;
    float* kb   = w + off; off += (size_t)Nn * Dd;
    float* vb   = w + off; off += (size_t)Nn * Dd;
    float* att  = w + off; off += (size_t)Nn * Dd;
    float* a2   = w + off; off += (size_t)Nn * Dd;
    float* hb   = w + off; off += (size_t)Nn * FFf;
    float* sco  = w + off; off += (size_t)Ee * Hh;
    float* eeb  = w + off; off += (size_t)Ee * Hh;
    float* smax = w + off; off += (size_t)Nn * Hh;
    float* den  = w + off; off += (size_t)Nn * Hh;
    float* csum = w + off; off += Dd;
    float* csq  = w + off; off += Dd;

    (void)hipMemcpyAsync(xb, x_in, (size_t)Nn * Dd * sizeof(float),
                         hipMemcpyDeviceToDevice, stream);

    const int EH = Ee * Hh, NH = Nn * Hh, ND = Nn * Dd, ED = Ee * Dd;
    dim3 b256(256);
    auto blocks = [](int n) { return dim3((unsigned)((n + 255) / 256)); };

    for (int i = 0; i < Ll; ++i) {
        const float* Wqi = Wq + (size_t)i * Dd * Dd;
        const float* Wki = Wk + (size_t)i * Dd * Dd;
        const float* Wvi = Wv + (size_t)i * Dd * Dd;
        const float* Woi = Wo + (size_t)i * Dd * Dd;
        const float* W1i = W1 + (size_t)i * Dd * FFf;
        const float* W2i = W2 + (size_t)i * FFf * Dd;

        // Q,K,V projections (WMMA)
        launch_gemm(xb, Wqi, nullptr, nullptr, qb, Nn, Dd, Dd, 0, stream);
        launch_gemm(xb, Wki, nullptr, nullptr, kb, Nn, Dd, Dd, 0, stream);
        launch_gemm(xb, Wvi, nullptr, nullptr, vb, Nn, Dd, Dd, 0, stream);

        // edge-softmax attention
        hipLaunchKernelGGL(k_scores, blocks(EH), b256, 0, stream, qb, kb, src, dst, sco);
        hipLaunchKernelGGL(k_fill, blocks(NH), b256, 0, stream, smax, -INFINITY, NH);
        hipLaunchKernelGGL(k_segmax, blocks(EH), b256, 0, stream, sco, dst, smax);
        hipLaunchKernelGGL(k_fixmax_zeroden, blocks(NH), b256, 0, stream, smax, den);
        hipLaunchKernelGGL(k_expsum, blocks(EH), b256, 0, stream, sco, dst, smax, eeb, den);
        hipLaunchKernelGGL(k_fill, blocks(ND), b256, 0, stream, att, 0.0f, ND);
        hipLaunchKernelGGL(k_aggregate, blocks(ED), b256, 0, stream,
                           eeb, den, vb, src, dst, att);

        // output projection + residual, then BN1
        launch_gemm(att, Woi, nullptr, xb, a2, Nn, Dd, Dd, 4, stream);
        hipLaunchKernelGGL(k_fill, dim3(1), b256, 0, stream, csum, 0.0f, 2 * Dd);
        hipLaunchKernelGGL(k_bn_stats, dim3(512), dim3(128), 0, stream, a2, csum, csq);
        hipLaunchKernelGGL(k_bn_apply, blocks(ND), b256, 0, stream,
                           a2, csum, csq, g1 + (size_t)i * Dd, bn1 + (size_t)i * Dd, xb);

        // FFN: relu(x@W1+b1) @ W2 + b2 + x, then BN2
        launch_gemm(xb, W1i, bf1 + (size_t)i * FFf, nullptr, hb, Nn, Dd, FFf, 1 | 2, stream);
        launch_gemm(hb, W2i, bf2 + (size_t)i * Dd, xb, a2, Nn, FFf, Dd, 1 | 4, stream);
        hipLaunchKernelGGL(k_fill, dim3(1), b256, 0, stream, csum, 0.0f, 2 * Dd);
        hipLaunchKernelGGL(k_bn_stats, dim3(512), dim3(128), 0, stream, a2, csum, csq);
        hipLaunchKernelGGL(k_bn_apply, blocks(ND), b256, 0, stream,
                           a2, csum, csq, g2 + (size_t)i * Dd, bn2 + (size_t)i * Dd, xb);
    }

    hipLaunchKernelGGL(k_head, blocks(Nn), b256, 0, stream, xb, Wp, bp, out);
}
